// Model_73332271612125
// MI455X (gfx1250) — matrix operand order repaired and checked by reading, not compiled, not run on hardware
//
#include <hip/hip_runtime.h>
#include <math.h>
#include <stdint.h>

// ---------------------------------------------------------------------------
// Types for CDNA5 WMMA (wave32) + TDM descriptors
// ---------------------------------------------------------------------------
typedef __attribute__((ext_vector_type(16))) __bf16 v16bf;
typedef __attribute__((ext_vector_type(8)))  float  v8f;
typedef __attribute__((ext_vector_type(4)))  unsigned int u32x4;
typedef __attribute__((ext_vector_type(8)))  int i32x8;
typedef __attribute__((ext_vector_type(4)))  int i32x4;

union ABfrag {
    v16bf          v;
    unsigned short u[16];
    unsigned int   d[8];
};

static __device__ __forceinline__ unsigned short f32_to_bf16(float f) {
    unsigned int x = __float_as_uint(f);
    unsigned int r = (x + 0x7FFFu + ((x >> 16) & 1u)) >> 16;   // RNE
    return (unsigned short)r;
}

// ---------------------------------------------------------------------------
// TDM: issue a 2D bf16 tile load (tile0 x tile1 elements) from a row-major
// tensor (row stride = stride0 elements, remaining dims td0 x td1 for OOB
// zero-fill) into LDS at byte offset ldsOff.  D# layout per ISA 8.3/8.4.
// ---------------------------------------------------------------------------
static __device__ __forceinline__ void tdm_load_2d(unsigned ldsOff,
                                                   const void* gptr,
                                                   int td0, int td1,
                                                   int tile0, int tile1,
                                                   int stride0)
{
    unsigned long long ga = (unsigned long long)(uintptr_t)gptr;
    u32x4 g0 = {0u, 0u, 0u, 0u};
    g0[0] = 1u;                                       // count=1, user desc
    g0[1] = ldsOff;                                   // lds_addr
    g0[2] = (unsigned)ga;                             // global_addr[31:0]
    g0[3] = (unsigned)((ga >> 32) & 0x01FFFFFFu)      // global_addr[56:32]
          | (2u << 30);                               // type=2 ("image")

    i32x8 g1 = {0, 0, 0, 0, 0, 0, 0, 0};
    unsigned utd0 = (unsigned)td0, utd1 = (unsigned)td1;
    g1[0] = 0x10000;                                  // data_size=1 (2 bytes)
    g1[1] = (int)((utd0 & 0xFFFFu) << 16);            // tensor_dim0[15:0]
    g1[2] = (int)((utd0 >> 16) | ((utd1 & 0xFFFFu) << 16));
    g1[3] = (int)((utd1 >> 16) | ((unsigned)tile0 << 16));  // tile_dim0
    g1[4] = (int)(unsigned)tile1;                     // tile_dim1 (tile_dim2=0)
    g1[5] = (int)(unsigned)stride0;                   // tensor_dim0_stride lo
    g1[6] = 0;                                        // stride hi + dim1_stride
    g1[7] = 0;

    i32x4 z4 = {0, 0, 0, 0};
#if __clang_major__ >= 23
    i32x8 z8 = {0, 0, 0, 0, 0, 0, 0, 0};
    __builtin_amdgcn_tensor_load_to_lds(g0, g1, z4, z4, z8, 0);
#else
    __builtin_amdgcn_tensor_load_to_lds(g0, g1, z4, z4, 0);
#endif
}

// ---------------------------------------------------------------------------
// Implicit-GEMM convolution (stride 1, SAME, odd kernel), NCHW fp32 in/out.
//   D[co][m] = sum_k Wbf[co][k] * patch[k][m]
// Block = 8 waves = 4 co-tiles x 2 pixel-tiles (64ch x 32px).
//  - A (weights): 64x32 bf16 tile async-copied to LDS each chunk
//    (global_load_async_to_lds_b32, ASYNCcnt), shared by 2 waves per co-tile.
//  - B (patches): gathered+converted cooperatively into LDS, shared by 4
//    waves per pixel-tile; fragments read as 2x ds_load_b128.
// ---------------------------------------------------------------------------
__global__ __launch_bounds__(256)
void conv_wmma_kernel(const float* __restrict__ x,
                      const unsigned short* __restrict__ wbf,
                      const float* __restrict__ bias,
                      float* __restrict__ y,
                      int N, int Cin, int H, int W, int Co,
                      int ks, int Kreal, int Kpad, int relu)
{
    const int pad = ks >> 1;
    const int KK  = ks * ks;
    const int M   = H * W;
    const int mt  = (M + 15) >> 4;     // 16-pixel tiles
    const int mtB = (mt + 1) >> 1;     // block covers 2 pixel tiles
    const int cotB = Co >> 6;          // block covers 4 co-tiles (64 ch)

    int blk = blockIdx.x;              // = ((n*cotB)+cb)*mtB + mb
    const int mb = blk % mtB; blk /= mtB;
    const int cb = blk % cotB;
    const int n  = blk / cotB;
    const int m0_blk = mb << 5;        // 32 pixels
    const int co_blk = cb << 6;        // 64 channels

    const int t    = threadIdx.x;
    const int wv   = t >> 5;           // wave 0..7
    const int l    = t & 31;
    const int half = l >> 4;
    const int l16  = l & 15;
    const int kb8  = half ? 8 : 0;
    const int kb16 = half << 4;

    const int co_base = co_blk + (wv & 3) * 16;   // wave's co tile
    const int mtl     = wv >> 2;                  // wave's pixel tile (0/1)

    const float* xn = x + (size_t)n * Cin * H * W;

    __shared__ unsigned short ldsB[2 * 16 * 32];  // patches [mtile][m][k] 2KB
    __shared__ unsigned short ldsA[64 * 32];      // weights [row][k]     4KB

    // B staging roles: thread t handles LDS elements t, t+256, t+512, t+768.
    int se_oh[4], se_ow[4];
    #pragma unroll
    for (int i = 0; i < 4; i++) {
        int e    = t + (i << 8);       // 0..1023
        int mtl2 = e >> 9;
        int m_l  = (e >> 5) & 15;
        int m    = m0_blk + mtl2 * 16 + m_l;
        se_oh[i] = m / W;
        se_ow[i] = m - se_oh[i] * W;
    }
    const unsigned ldsA0 = (unsigned)(uintptr_t)&ldsA[0];

    v8f acc = {};
    for (int k0 = 0; k0 < Kpad; k0 += 32) {
        __syncthreads();
        // ---- async-copy weight tile: 1024 dwords, 4 per thread ----
        #pragma unroll
        for (int i = 0; i < 4; i++) {
            int e = t + (i << 8);
            unsigned gOff = (unsigned)(((unsigned)(co_blk + (e >> 4)) * (unsigned)Kpad
                                        + (unsigned)k0 + (unsigned)((e & 15) << 1)) << 1);
            unsigned ldsOff = ldsA0 + ((unsigned)e << 2);
            asm volatile("global_load_async_to_lds_b32 %0, %1, %2"
                         :: "v"(ldsOff), "v"(gOff), "s"(wbf) : "memory");
        }
        // ---- cooperative patch staging (gather + bf16 convert) ----
        #pragma unroll
        for (int i = 0; i < 4; i++) {
            int e  = t + (i << 8);
            int kk = k0 + (e & 31);
            float v = 0.0f;
            if (kk < Kreal && se_oh[i] < H) {
                int ci = kk / KK;
                int r  = kk - ci * KK;
                int kh = r / ks;
                int kw = r - kh * ks;
                int ih = se_oh[i] + kh - pad;
                int iw = se_ow[i] + kw - pad;
                if (ih >= 0 && ih < H && iw >= 0 && iw < W)
                    v = xn[((size_t)ci * H + ih) * W + iw];
            }
            ldsB[e] = f32_to_bf16(v);
        }
        asm volatile("s_wait_asynccnt 0x0" ::: "memory");
        __syncthreads();

        // ---- A fragment from LDS: two 16B runs per lane ----
        ABfrag a, b2;
        const unsigned short* arow = &ldsA[((wv & 3) * 16 + l16) * 32];
        uint4 a0 = *(const uint4*)(arow + kb8);
        uint4 a1 = *(const uint4*)(arow + kb8 + 16);
        a.d[0] = a0.x; a.d[1] = a0.y; a.d[2] = a0.z; a.d[3] = a0.w;
        a.d[4] = a1.x; a.d[5] = a1.y; a.d[6] = a1.z; a.d[7] = a1.w;

        // ---- B fragment from LDS: two ds_load_b128 per lane ----
        const uint4* bp = (const uint4*)&ldsB[(mtl * 16 + l16) * 32 + kb8];
        uint4 b0 = bp[0];
        uint4 b1 = bp[2];
        b2.d[0] = b0.x; b2.d[1] = b0.y; b2.d[2] = b0.z; b2.d[3] = b0.w;
        b2.d[4] = b1.x; b2.d[5] = b1.y; b2.d[6] = b1.z; b2.d[7] = b1.w;

        acc = __builtin_amdgcn_wmma_f32_16x16x32_bf16(
                  false, a.v, false, b2.v, (short)0, acc, false, false);
    }

    const int m = m0_blk + mtl * 16 + l16;
    if (m < M) {
        float* yn = y + (size_t)n * Co * M;
        #pragma unroll
        for (int v2 = 0; v2 < 8; v2++) {
            const int co = co_base + (half ? v2 + 8 : v2);
            float r = acc[v2] + bias[co];
            if (relu) r = fmaxf(r, 0.0f);
            yn[(size_t)co * M + m] = r;
        }
    }
}

// ---------------------------------------------------------------------------
// GEMM-NT on bf16 rows:  C[i][j] = sum_k A[i][k] * B[j][k]
// A: [M][K], B: [N][K] bf16 row-major, K % 128 == 0.
// Block = 8 waves x 16 A-rows, sharing a 64-column B panel.  B panels
// (64 x 128 bf16 = 16KB) are double-buffered in LDS and fetched by the
// Tensor Data Mover (tensor_load_to_lds, TENSORcnt), issued by wave 0.
// Each wave: 1 A-fragment (2x global b128) feeds 4 WMMAs per k-chunk.
// ---------------------------------------------------------------------------
#define PANEL_K 128

__global__ __launch_bounds__(256)
void gemm_nt_wmma_kernel(const unsigned short* __restrict__ A,
                         const unsigned short* __restrict__ B,
                         float* __restrict__ C,
                         int M, int N, int K, int ldc)
{
    const int jbCount = (N + 63) >> 6;
    int blk = blockIdx.x;              // = ib*jbCount + jb
    const int jb = blk % jbCount;
    const int ib = blk / jbCount;
    const int j0_blk = jb << 6;

    const int t = threadIdx.x, wv = t >> 5, l = t & 31;
    const int half = l >> 4, l16 = l & 15;
    const int kb8  = half ? 8 : 0;
    const int kb16 = half << 4;

    const int i0   = ((ib << 3) + wv) << 4;
    const int arow = min(i0 + l16, M - 1);
    const unsigned int* ap = (const unsigned int*)(A + (size_t)arow * K);

    __shared__ unsigned short ldsB[2][64 * PANEL_K];   // 2 x 16KB

    const int P = K / PANEL_K;
    const int wv0 = __builtin_amdgcn_readfirstlane(wv);   // uniform branch

    if (wv0 == 0) {
        tdm_load_2d((unsigned)(uintptr_t)&ldsB[0][0],
                    B + (size_t)j0_blk * K,
                    K, N - j0_blk, PANEL_K, 64, K);
    }

    v8f acc[4] = {};
    for (int p = 0; p < P; p++) {
        if (wv0 == 0) {
            int pn = min(p + 1, P - 1);     // redundant re-issue on last panel
            tdm_load_2d((unsigned)(uintptr_t)&ldsB[(p + 1) & 1][0],
                        B + ((size_t)j0_blk * K + (size_t)pn * PANEL_K),
                        K - pn * PANEL_K, N - j0_blk, PANEL_K, 64, K);
            __builtin_amdgcn_s_wait_tensorcnt(1);   // panel p landed
        }
        __syncthreads();

        const unsigned short* bb = &ldsB[p & 1][0];
        const int kbase = p * PANEL_K;
        #pragma unroll
        for (int kp = 0; kp < PANEL_K; kp += 32) {
            ABfrag a;
            uint4 a0 = *(const uint4*)(ap + ((kbase + kp + kb8) >> 1));
            uint4 a1 = *(const uint4*)(ap + ((kbase + kp + kb8 + 16) >> 1));
            a.d[0] = a0.x; a.d[1] = a0.y; a.d[2] = a0.z; a.d[3] = a0.w;
            a.d[4] = a1.x; a.d[5] = a1.y; a.d[6] = a1.z; a.d[7] = a1.w;

            #pragma unroll
            for (int jl = 0; jl < 4; jl++) {
                ABfrag b2;
                const uint4* bp =
                    (const uint4*)&bb[(jl * 16 + l16) * PANEL_K + kp + kb8];
                uint4 b0 = bp[0];
                uint4 b1 = bp[2];
                b2.d[0] = b0.x; b2.d[1] = b0.y; b2.d[2] = b0.z; b2.d[3] = b0.w;
                b2.d[4] = b1.x; b2.d[5] = b1.y; b2.d[6] = b1.z; b2.d[7] = b1.w;

                acc[jl] = __builtin_amdgcn_wmma_f32_16x16x32_bf16(
                              false, a.v, false, b2.v, (short)0, acc[jl],
                              false, false);
            }
        }
        __syncthreads();
    }

    #pragma unroll
    for (int jl = 0; jl < 4; jl++) {
        const int j = j0_blk + jl * 16 + l16;
        if (j < N) {
            #pragma unroll
            for (int v2 = 0; v2 < 8; v2++) {
                const int i = i0 + (half ? v2 + 8 : v2);
                if (i < M) C[(size_t)i * ldc + j] = acc[jl][v2];
            }
        }
    }
}

// ---------------------------------------------------------------------------
// fp32 -> bf16 weight repack with K padding to 32
// ---------------------------------------------------------------------------
__global__ void convert_w_kernel(const float* __restrict__ w,
                                 unsigned short* __restrict__ wbf,
                                 int Co, int Kreal, int Kpad)
{
    long long idx = (long long)blockIdx.x * blockDim.x + threadIdx.x;
    if (idx >= (long long)Co * Kpad) return;
    int co = (int)(idx / Kpad);
    int k  = (int)(idx - (long long)co * Kpad);
    float v = (k < Kreal) ? w[(size_t)co * Kreal + k] : 0.0f;
    wbf[idx] = f32_to_bf16(v);
}

// ---------------------------------------------------------------------------
// 2x2 maxpool, NCHW
// ---------------------------------------------------------------------------
__global__ void maxpool_kernel(const float* __restrict__ x, float* __restrict__ y,
                               int N, int C, int H, int W)
{
    const int OH = H >> 1, OW = W >> 1;
    long long idx = (long long)blockIdx.x * blockDim.x + threadIdx.x;
    if (idx >= (long long)N * C * OH * OW) return;
    int ow = (int)(idx % OW); long long q = idx / OW;
    int oh = (int)(q % OH);   q /= OH;
    int c  = (int)(q % C);    int n = (int)(q / C);
    const float* p = x + (((size_t)(n * C + c) * H + 2 * oh) * W + 2 * ow);
    y[idx] = fmaxf(fmaxf(p[0], p[1]), fmaxf(p[W], p[W + 1]));
}

// ---------------------------------------------------------------------------
// Per-pixel channel sum (feat_norm) + 1x1 conv dot (logits) + sigmoid.
// ---------------------------------------------------------------------------
__global__ void feat_post_kernel(const float* __restrict__ feat,
                                 const float* __restrict__ c1w,
                                 const float* __restrict__ c1b,
                                 float* __restrict__ rnorm,
                                 float* __restrict__ logits,
                                 float* __restrict__ ssal)
{
    const int m  = blockIdx.x;          // 0..783
    const int n  = m / 196;
    const int hw = m - n * 196;
    const float* f = feat + (size_t)n * 4096 * 196 + hw;

    float s = 0.0f, d = 0.0f;
    for (int c = threadIdx.x; c < 4096; c += blockDim.x) {
        float v = f[(size_t)c * 196];
        s += v;
        d += v * c1w[c];
    }
    __shared__ float sh0[256];
    __shared__ float sh1[256];
    sh0[threadIdx.x] = s; sh1[threadIdx.x] = d;
    __syncthreads();
    for (int o = 128; o > 0; o >>= 1) {
        if (threadIdx.x < o) {
            sh0[threadIdx.x] += sh0[threadIdx.x + o];
            sh1[threadIdx.x] += sh1[threadIdx.x + o];
        }
        __syncthreads();
    }
    if (threadIdx.x == 0) {
        rnorm[m] = rsqrtf(sh0[0] + 0.01f);
        float lg = sh1[0] + c1b[0];
        logits[m] = lg;
        ssal[m] = 1.0f / (1.0f + expf(-lg));
    }
}

// ---------------------------------------------------------------------------
// Build bf16 row-major feature matrices.
// ---------------------------------------------------------------------------
__global__ void make_fw_kernel(const float* __restrict__ feat,
                               const float* __restrict__ rnorm,
                               const float* __restrict__ ssal,
                               unsigned short* __restrict__ Fbf,
                               unsigned short* __restrict__ Wbf)
{
    long long idx = (long long)blockIdx.x * blockDim.x + threadIdx.x;
    if (idx >= 784LL * 4096) return;
    int m = (int)(idx >> 12);
    int c = (int)(idx & 4095);
    int n = m / 196, hw = m - n * 196;
    float fn = feat[((size_t)(n * 4096 + c)) * 196 + hw] * rnorm[m];
    Fbf[idx] = f32_to_bf16(fn);
    Wbf[idx] = f32_to_bf16(fn * ssal[m]);
}

// ---------------------------------------------------------------------------
// x32 bilinear upsample (half-pixel centers, edge clamp) + sigmoid
// ---------------------------------------------------------------------------
__global__ void upsample_sigmoid_kernel(const float* __restrict__ logits,
                                        float* __restrict__ out)
{
    long long idx = (long long)blockIdx.x * blockDim.x + threadIdx.x;
    if (idx >= 4LL * 448 * 448) return;
    int n = (int)(idx / (448 * 448));
    int r = (int)(idx - (long long)n * 448 * 448);
    int oy = r / 448, ox = r - oy * 448;
    float fy = (oy + 0.5f) * 0.03125f - 0.5f;
    float fx = (ox + 0.5f) * 0.03125f - 0.5f;
    fy = fminf(fmaxf(fy, 0.0f), 13.0f);
    fx = fminf(fmaxf(fx, 0.0f), 13.0f);
    int y0 = min((int)fy, 12), x0 = min((int)fx, 12);
    float wy = fy - y0, wx = fx - x0;
    const float* L = logits + n * 196;
    float v00 = L[y0 * 14 + x0],       v01 = L[y0 * 14 + x0 + 1];
    float v10 = L[(y0 + 1) * 14 + x0], v11 = L[(y0 + 1) * 14 + x0 + 1];
    float v = v00 * (1 - wy) * (1 - wx) + v01 * (1 - wy) * wx
            + v10 * wy * (1 - wx)       + v11 * wy * wx;
    out[idx] = 1.0f / (1.0f + expf(-v));
}

// ---------------------------------------------------------------------------
// sal_affinity + sal_diff
// ---------------------------------------------------------------------------
__global__ void sal_pair_kernel(const float* __restrict__ ssal,
                                float* __restrict__ oaf,
                                float* __restrict__ odiff)
{
    long long idx = (long long)blockIdx.x * blockDim.x + threadIdx.x;
    if (idx >= 784LL * 784) return;
    int i = (int)(idx / 784);
    int j = (int)(idx - (long long)i * 784);
    float a = ssal[i], b = ssal[j];
    oaf[idx] = a * b;
    float d = a - b;
    odiff[idx] = d * d;
}

// ---------------------------------------------------------------------------
// Median of 38416 elems per pair: 32-step binary search on ordered uint keys.
// ---------------------------------------------------------------------------
__global__ void median_kernel(const float* __restrict__ fourD, float* __restrict__ med)
{
    const int p = blockIdx.x;
    const float* t = fourD + (size_t)p * 38416;
    const int n = 38416;
    __shared__ unsigned int cnt[256];

    float vals[2];
    for (int s = 0; s < 2; s++) {
        const unsigned int k = 19207u + (unsigned int)s;   // 0-based rank
        unsigned int lo = 0u, hi = 0xFFFFFFFFu;
        while (lo < hi) {
            unsigned int mid = lo + ((hi - lo) >> 1);
            unsigned int c = 0;
            for (int e = threadIdx.x; e < n; e += blockDim.x) {
                unsigned int u = __float_as_uint(t[e]);
                unsigned int key = (u & 0x80000000u) ? ~u : (u | 0x80000000u);
                c += (key <= mid) ? 1u : 0u;
            }
            cnt[threadIdx.x] = c;
            __syncthreads();
            for (int o = 128; o > 0; o >>= 1) {
                if (threadIdx.x < o) cnt[threadIdx.x] += cnt[threadIdx.x + o];
                __syncthreads();
            }
            unsigned int total = cnt[0];
            __syncthreads();
            if (total >= k + 1u) hi = mid; else lo = mid + 1u;
        }
        unsigned int key = lo;
        unsigned int u = (key & 0x80000000u) ? (key ^ 0x80000000u) : ~key;
        vals[s] = __uint_as_float(u);
    }
    if (threadIdx.x == 0) med[p] = 0.5f * (vals[0] + vals[1]);
}

// ---------------------------------------------------------------------------
// Co-peak: 3^4 max filter (symmetric pad==clamp for pad=1), mask, mean.
// ---------------------------------------------------------------------------
__global__ void zero_accum_kernel(float* __restrict__ accum)
{
    if (threadIdx.x < 4) accum[threadIdx.x] = 0.0f;
}

__global__ void copeak_kernel(const float* __restrict__ fourD,
                              const float* __restrict__ med,
                              float* __restrict__ accum)
{
    int idx = blockIdx.x * blockDim.x + threadIdx.x;
    if (idx >= 2 * 38416) return;
    int p = idx / 38416;
    int r = idx - p * 38416;
    int d = r % 14; int q = r / 14;
    int c = q % 14; q /= 14;
    int b = q % 14; int a = q / 14;
    const float* T = fourD + (size_t)p * 38416;
    float v = T[r];
    float mx = -3.402823e38f;
    for (int da = -1; da <= 1; da++) {
        int aa = min(max(a + da, 0), 13);
        for (int db = -1; db <= 1; db++) {
            int bb = min(max(b + db, 0), 13);
            for (int dc = -1; dc <= 1; dc++) {
                int cc = min(max(c + dc, 0), 13);
                for (int dd = -1; dd <= 1; dd++) {
                    int de = min(max(d + dd, 0), 13);
                    mx = fmaxf(mx, T[((aa * 14 + bb) * 14 + cc) * 14 + de]);
                }
            }
        }
    }
    if (v == mx && v >= med[p]) {
        atomicAdd(&accum[2 * p], v);
        atomicAdd(&accum[2 * p + 1], 1.0f);
    }
}

__global__ void copeak_final_kernel(const float* __restrict__ accum,
                                    float* __restrict__ out)
{
    if (threadIdx.x < 2)
        out[threadIdx.x] = accum[2 * threadIdx.x] / accum[2 * threadIdx.x + 1];
}

// ---------------------------------------------------------------------------
// Host driver.
// Input order (dict flattened): x, vgg_w[0..12], vgg_b[0..12],
//                               conv6_w, conv6_b, conv1_w, conv1_b
// ---------------------------------------------------------------------------
extern "C" void kernel_launch(void* const* d_in, const int* in_sizes, int n_in,
                              void* d_out, int out_size, void* d_ws, size_t ws_size,
                              hipStream_t stream)
{
    (void)in_sizes; (void)n_in; (void)out_size; (void)ws_size;

    const float* x = (const float*)d_in[0];
    const float* vw[13]; const float* vb[13];
    for (int i = 0; i < 13; i++) {
        vw[i] = (const float*)d_in[1 + i];
        vb[i] = (const float*)d_in[14 + i];
    }
    const float* c6w = (const float*)d_in[27];
    const float* c6b = (const float*)d_in[28];
    const float* c1w = (const float*)d_in[29];
    const float* c1b = (const float*)d_in[30];

    // ---- workspace carve ----
    char* base = (char*)d_ws;
    size_t off = 0;
    auto carve = [&](size_t bytes) -> void* {
        off = (off + 255) & ~(size_t)255;
        void* r = base + off;
        off += bytes;
        return r;
    };
    float*          bufA   = (float*)carve((size_t)4 * 64 * 448 * 448 * sizeof(float));
    float*          bufB   = (float*)carve((size_t)4 * 64 * 448 * 448 * sizeof(float));
    unsigned short* wbf    = (unsigned short*)carve((size_t)4096 * 25088 * sizeof(unsigned short));
    unsigned short* Fbf    = (unsigned short*)carve((size_t)784 * 4096 * sizeof(unsigned short));
    unsigned short* Wbf    = (unsigned short*)carve((size_t)784 * 4096 * sizeof(unsigned short));
    float*          fourD  = (float*)carve((size_t)2 * 38416 * sizeof(float));
    float*          rnorm  = (float*)carve(784 * sizeof(float));
    float*          logitsB= (float*)carve(784 * sizeof(float));
    float*          ssal   = (float*)carve(784 * sizeof(float));
    float*          med    = (float*)carve(2 * sizeof(float));
    float*          accum  = (float*)carve(4 * sizeof(float));

    float* o_sal = (float*)d_out;
    float* o_pix = o_sal + 802816;
    float* o_saf = o_pix + 614656;
    float* o_sdf = o_saf + 614656;
    float* o_cpk = o_sdf + 614656;

    // ---- VGG16 backbone ----
    const int cfg[18] = {64,64,-1,128,128,-1,256,256,256,-1,
                         512,512,512,-1,512,512,512,-1};
    const float* cur = x;
    float* nxt = bufA;
    int C = 3, H = 448, W = 448, li = 0;

    for (int ci = 0; ci < 18; ci++) {
        if (cfg[ci] < 0) {
            int OH = H >> 1, OW = W >> 1;
            long long tot = 4LL * C * OH * OW;
            maxpool_kernel<<<(int)((tot + 255) / 256), 256, 0, stream>>>(
                cur, nxt, 4, C, H, W);
            cur = nxt; nxt = (nxt == bufA) ? bufB : bufA;
            H = OH; W = OW;
        } else {
            int Co = cfg[ci], Kreal = C * 9, Kpad = ((Kreal + 31) / 32) * 32;
            long long wtot = (long long)Co * Kpad;
            convert_w_kernel<<<(int)((wtot + 255) / 256), 256, 0, stream>>>(
                vw[li], wbf, Co, Kreal, Kpad);
            int M = H * W, mt = (M + 15) / 16;
            int mtB = (mt + 1) / 2, cotB = Co / 64;
            int blocks = 4 * cotB * mtB;
            conv_wmma_kernel<<<blocks, 256, 0, stream>>>(
                cur, wbf, vb[li], nxt, 4, C, H, W, Co, 3, Kreal, Kpad, 1);
            cur = nxt; nxt = (nxt == bufA) ? bufB : bufA;
            C = Co; li++;
        }
    }

    // ---- conv6: 7x7, 512 -> 4096 @ 14x14 ----
    {
        int Co = 4096, Kreal = 512 * 49, Kpad = Kreal;   // 25088 % 32 == 0
        long long wtot = (long long)Co * Kpad;
        convert_w_kernel<<<(int)((wtot + 255) / 256), 256, 0, stream>>>(
            c6w, wbf, Co, Kreal, Kpad);
        int mt = (196 + 15) / 16;                 // 13
        int mtB = (mt + 1) / 2, cotB = Co / 64;   // 7, 64
        int blocks = 4 * cotB * mtB;
        conv_wmma_kernel<<<blocks, 256, 0, stream>>>(
            cur, wbf, c6b, nxt, 4, 512, 14, 14, 4096, 7, Kreal, Kpad, 1);
        cur = nxt; nxt = (nxt == bufA) ? bufB : bufA;
    }
    const float* feat = cur;   // (4,4096,14,14) fp32, ReLU'd

    // ---- norm, 1x1 conv, sigmoid ----
    feat_post_kernel<<<784, 256, 0, stream>>>(feat, c1w, c1b, rnorm, logitsB, ssal);
    make_fw_kernel<<<(784 * 4096) / 256, 256, 0, stream>>>(feat, rnorm, ssal, Fbf, Wbf);

    // ---- salmap (x32 bilinear + sigmoid) ----
    upsample_sigmoid_kernel<<<(4 * 448 * 448) / 256, 256, 0, stream>>>(logitsB, o_sal);

    // ---- sal_affinity + sal_diff ----
    sal_pair_kernel<<<(784 * 784 + 255) / 256, 256, 0, stream>>>(ssal, o_saf, o_sdf);

    // ---- pixel_affinity = F F^T (784x784x4096) ----
    {
        int mt = (784 + 15) / 16;                  // 49
        int blocks = ((784 + 63) / 64) * ((mt + 7) / 8);   // 13 * 7
        gemm_nt_wmma_kernel<<<blocks, 256, 0, stream>>>(
            Fbf, Fbf, o_pix, 784, 784, 4096, 784);
    }

    // ---- fourD einsum per pair (196x196x4096) ----
    for (int p = 0; p < 2; p++) {
        int mt = (196 + 15) / 16;                  // 13
        int blocks = ((196 + 63) / 64) * ((mt + 7) / 8);   // 4 * 2
        gemm_nt_wmma_kernel<<<blocks, 256, 0, stream>>>(
            Wbf + (size_t)p * 196 * 4096,
            Wbf + (size_t)(2 + p) * 196 * 4096,
            fourD + (size_t)p * 38416, 196, 196, 4096, 196);
    }

    // ---- co_peak ----
    median_kernel<<<2, 256, 0, stream>>>(fourD, med);
    zero_accum_kernel<<<1, 32, 0, stream>>>(accum);
    copeak_kernel<<<(2 * 38416 + 255) / 256, 256, 0, stream>>>(fourD, med, accum);
    copeak_final_kernel<<<1, 32, 0, stream>>>(accum, o_cpk);
}
